// PointsFusion_5291399708686
// MI455X (gfx1250) — matrix operand order repaired and checked by reading, not compiled, not run on hardware
//
#include <hip/hip_runtime.h>
#include <hip/hip_fp16.h>
#include <math.h>

// ---------------------------------------------------------------------------
// PointsFusion for MI455X (gfx1250, wave32, WMMA, async global->LDS).
// Activations are SAMPLE-MAJOR f16 [S][C]: B-fragments and D-fragments are
// layout-native (lane == sample), so no LDS transpose is needed anywhere.
//   1) knn_feats_kernel : top-32 NN vs p1 and p2 -> feats[S][4], grouped[S][3]
//   2) conv1_wmma       : 4->64 WMMA conv (K padded to 32), raw acc + stats
//   3) finalize_stats   : scale/shift from batch stats (conv bias cancels in BN)
//   4) gemm_bn_relu     : WMMA 16x16x32 f16; packed-f16 BN+relu on load;
//                         double-buffered async global->LDS staging.
//   5) fuse_kernel      : BN3+relu, channel max, softmax over 64 nbrs, weighted sum
// ---------------------------------------------------------------------------

typedef __attribute__((ext_vector_type(16))) _Float16 v16h;
typedef __attribute__((ext_vector_type(8)))  _Float16 v8h;
typedef __attribute__((ext_vector_type(8)))  float    v8f;

#define BB 4
#define NN 4096
#define KK 32
#define K2 64
#define SS (BB * NN * K2)      // 1,048,576 samples
#define GT 128                 // samples per block macro-tile
#define BN_EPS 1e-3f

// ---------------------------------------------------------------- KNN -------
__global__ __launch_bounds__(128)
void knn_feats_kernel(const float* __restrict__ pts1, const float* __restrict__ pts2,
                      float* __restrict__ feats, float* __restrict__ grouped) {
    __shared__ float sp[NN * 3];              // 48KB candidate buffer
    const int b = blockIdx.y;
    const int n = blockIdx.x * 128 + threadIdx.x;
    const float* p1 = pts1 + (long)b * 3 * NN;
    const float* p2 = pts2 + (long)b * 3 * NN;
    const float qx = p1[0 * NN + n], qy = p1[1 * NN + n], qz = p1[2 * NN + n];

    for (int pass = 0; pass < 2; ++pass) {
        const float* src = pass ? p2 : p1;
        __syncthreads();                      // protect sp from previous pass readers
        for (int j = threadIdx.x; j < NN; j += 128) {
            sp[j * 3 + 0] = src[0 * NN + j];
            sp[j * 3 + 1] = src[1 * NN + j];
            sp[j * 3 + 2] = src[2 * NN + j];
        }
        __syncthreads();

        float bd[KK]; int bi[KK];
        #pragma unroll
        for (int j = 0; j < KK; ++j) { bd[j] = 3.4e38f; bi[j] = 0; }

        #pragma unroll 4
        for (int c = 0; c < NN; ++c) {
            const float dx = sp[c * 3 + 0] - qx;
            const float dy = sp[c * 3 + 1] - qy;
            const float dz = sp[c * 3 + 2] - qz;
            const float d2 = dx * dx + dy * dy + dz * dz;
            if (d2 < bd[KK - 1]) {            // sorted-insert (unrolled bubble)
                float v = d2; int vi = c;
                #pragma unroll
                for (int j = 0; j < KK; ++j) {
                    if (v < bd[j]) {
                        float tv = bd[j]; bd[j] = v;  v  = tv;
                        int   ti = bi[j]; bi[j] = vi; vi = ti;
                    }
                }
            }
        }

        const long base = ((long)(b * NN + n)) * K2 + pass * KK;
        for (int j = 0; j < KK; ++j) {
            const int c = bi[j];
            const float nx = sp[c * 3 + 0], ny = sp[c * 3 + 1], nz = sp[c * 3 + 2];
            const float rx = nx - qx, ry = ny - qy, rz = nz - qz;
            const float dist = sqrtf(rx * rx + ry * ry + rz * rz);
            const long o = base + j;
            feats[o * 4 + 0] = rx; feats[o * 4 + 1] = ry;
            feats[o * 4 + 2] = rz; feats[o * 4 + 3] = dist;
            grouped[o * 3 + 0] = nx; grouped[o * 3 + 1] = ny; grouped[o * 3 + 2] = nz;
        }
    }
}

// ----------------------------------------------------- layer 1 (WMMA) -------
// K padded 4 -> 32.  y1 is [S][64] f16, raw accumulator (bias cancels in BN).
__global__ __launch_bounds__(256)
void conv1_wmma_kernel(const float* __restrict__ feats, const float* __restrict__ W1,
                       _Float16* __restrict__ y1,
                       float* __restrict__ sumBuf, float* __restrict__ sqBuf) {
    __shared__ float ssum[64], ssq[64];
    const int t = threadIdx.x;
    if (t < 64) { ssum[t] = 0.f; ssq[t] = 0.f; }
    const int lane = t & 31, wib = t >> 5;
    const int lhi = lane >> 4, llo = lane & 15;
    const int j0 = wib * 16 + llo;

    v16h a[4];
    #pragma unroll
    for (int mt = 0; mt < 4; ++mt)
        #pragma unroll
        for (int h = 0; h < 16; ++h) {
            const int col = h + 16 * lhi;
            a[mt][h] = (col < 4) ? (_Float16)W1[(mt * 16 + llo) * 4 + col]
                                 : (_Float16)0.f;
        }

    float psum[4][8] = {}, psq[4][8] = {};
    __syncthreads();

    for (int tile = blockIdx.x; tile < SS / GT; tile += gridDim.x) {
        const long s0 = (long)tile * GT;
        const float4 x = ((const float4*)feats)[s0 + j0];   // coalesced b128

        v16h bf;
        #pragma unroll
        for (int h = 0; h < 16; ++h) {
            const int c = 16 * lhi + h;
            const float v = (c == 0) ? x.x : (c == 1) ? x.y : (c == 2) ? x.z
                          : (c == 3) ? x.w : 0.f;
            bf[h] = (_Float16)v;
        }

        v8f acc[4] = {{}, {}, {}, {}};
        #pragma unroll
        for (int mt = 0; mt < 4; ++mt)
            acc[mt] = __builtin_amdgcn_wmma_f32_16x16x32_f16(
                false, a[mt], false, bf, (short)0, acc[mt], false, false);

        // D is sample-major-native: 8 consecutive channels per mt -> one b128 store
        #pragma unroll
        for (int mt = 0; mt < 4; ++mt) {
            v8h o8;
            #pragma unroll
            for (int r = 0; r < 8; ++r) {
                const float y = acc[mt][r];
                psum[mt][r] += y; psq[mt][r] += y * y;
                o8[r] = (_Float16)y;
            }
            *(v8h*)(y1 + (s0 + j0) * 64 + mt * 16 + 8 * lhi) = o8;
        }
    }

    #pragma unroll
    for (int mt = 0; mt < 4; ++mt)
        #pragma unroll
        for (int r = 0; r < 8; ++r) {
            float ps = psum[mt][r], pq = psq[mt][r];
            #pragma unroll
            for (int m = 8; m >= 1; m >>= 1) { ps += __shfl_xor(ps, m, 32); pq += __shfl_xor(pq, m, 32); }
            if (llo == 0) {
                const int lc = mt * 16 + r + 8 * lhi;
                atomicAdd(&ssum[lc], ps); atomicAdd(&ssq[lc], pq);
            }
        }
    __syncthreads();
    if (t < 64) { atomicAdd(&sumBuf[t], ssum[t]); atomicAdd(&sqBuf[t], ssq[t]); }
}

// ------------------------------------------------------- BN finalization ----
__global__ void finalize_stats_kernel(const float* __restrict__ sum, const float* __restrict__ sq,
                                      const float* __restrict__ g, const float* __restrict__ be,
                                      float* __restrict__ scale, float* __restrict__ shift, int C) {
    const int c = blockIdx.x * blockDim.x + threadIdx.x;
    if (c >= C) return;
    const float inv = 1.0f / (float)SS;
    const float m = sum[c] * inv;
    const float v = sq[c] * inv - m * m;          // biased var (jnp.var default)
    const float r = rsqrtf(v + BN_EPS);
    const float sc = g[c] * r;
    scale[c] = sc;
    shift[c] = be[c] - m * sc;
}

// --------------------------------------------- WMMA GEMM + BN-in + stats ----
// Wave tile 64(M)x16(N)x64(K); block macro-tile = 128 samples = one flat 16KB
// global block (sample-major), staged with double-buffered async loads.
__global__ __launch_bounds__(256)
void gemm_bn_relu_kernel(const float* __restrict__ W,        // [CoutTot][64] row-major
                         const _Float16* __restrict__ Xin,   // [S][64] raw acc
                         const float* __restrict__ scaleIn,  // [64]
                         const float* __restrict__ shiftIn,  // [64]
                         _Float16* __restrict__ Yout,        // [S][CoutTot] raw acc
                         float* __restrict__ sumBuf, float* __restrict__ sqBuf,
                         int m_offset, int CoutTot) {
    __shared__ __align__(16) _Float16 sx[2][64 * GT];   // 2 x 16KB double buffer
    __shared__ float ssum[64], ssq[64];
    const int t = threadIdx.x;
    if (t < 64) { ssum[t] = 0.f; ssq[t] = 0.f; }
    const int lane = t & 31, wib = t >> 5;
    const int lhi = lane >> 4, llo = lane & 15;
    const int j0 = wib * 16 + llo;

    // Resident weight fragments (f32 -> f16): 8 x v16h = 64 VGPRs
    v16h a[4][2];
    #pragma unroll
    for (int mt = 0; mt < 4; ++mt)
        #pragma unroll
        for (int ks = 0; ks < 2; ++ks)
            #pragma unroll
            for (int h = 0; h < 16; ++h) {
                const int row = m_offset + mt * 16 + llo;
                const int col = h + 16 * lhi + 32 * ks;
                a[mt][ks][h] = (_Float16)W[row * 64 + col];
            }

    // Per-lane packed BN constants (channels per lane are tile-invariant)
    v16h sclh[2], shfh[2];
    #pragma unroll
    for (int ks = 0; ks < 2; ++ks)
        #pragma unroll
        for (int h = 0; h < 16; ++h) {
            const int c = 16 * lhi + h + 32 * ks;
            sclh[ks][h] = (_Float16)scaleIn[c];
            shfh[ks][h] = (_Float16)shiftIn[c];
        }

    float psum[4][8] = {}, psq[4][8] = {};
    const int numTiles = SS / GT;
    int cb = 0;

    // Preamble: stage first tile (contiguous 16KB), async b128 x4 per thread
    {
        const long s0 = (long)blockIdx.x * GT;
        #pragma unroll
        for (int i = 0; i < 4; ++i) {
            const int o = (t + 256 * i) * 8;              // in halves (16B chunks)
            const unsigned ldst = (unsigned)(uintptr_t)(&sx[0][o]);
            const void* gp = (const void*)(Xin + s0 * 64 + o);
            asm volatile("global_load_async_to_lds_b128 %0, %1, off"
                         :: "v"(ldst), "v"(gp) : "memory");
        }
        asm volatile("s_wait_asynccnt 0" ::: "memory");
    }
    __syncthreads();

    for (int tile = blockIdx.x; tile < numTiles; tile += gridDim.x) {
        const long s0 = (long)tile * GT;
        const int nextTile = tile + gridDim.x;

        // Prefetch next tile into the other buffer (async, no wait yet)
        if (nextTile < numTiles) {
            const long sn0 = (long)nextTile * GT;
            #pragma unroll
            for (int i = 0; i < 4; ++i) {
                const int o = (t + 256 * i) * 8;
                const unsigned ldst = (unsigned)(uintptr_t)(&sx[cb ^ 1][o]);
                const void* gp = (const void*)(Xin + sn0 * 64 + o);
                asm volatile("global_load_async_to_lds_b128 %0, %1, off"
                             :: "v"(ldst), "v"(gp) : "memory");
            }
        }

        // B fragments: one v16h LDS read per k-step; packed f16 BN + relu
        v8f acc[4] = {{}, {}, {}, {}};
        #pragma unroll
        for (int ks = 0; ks < 2; ++ks) {
            const v16h xv = *(const v16h*)(&sx[cb][j0 * 64 + 16 * lhi + 32 * ks]);
            v16h bf;
            #pragma unroll
            for (int h = 0; h < 16; ++h) {
                const _Float16 v = xv[h] * sclh[ks][h] + shfh[ks][h];
                bf[h] = v > (_Float16)0.f ? v : (_Float16)0.f;
            }
            #pragma unroll
            for (int mt = 0; mt < 4; ++mt)
                acc[mt] = __builtin_amdgcn_wmma_f32_16x16x32_f16(
                    false, a[mt][ks], false, bf, (short)0, acc[mt], false, false);
        }

        // Epilogue: register stats + direct b128 global stores (sample-major D)
        #pragma unroll
        for (int mt = 0; mt < 4; ++mt) {
            v8h o8;
            #pragma unroll
            for (int r = 0; r < 8; ++r) {
                const float y = acc[mt][r];
                psum[mt][r] += y; psq[mt][r] += y * y;
                o8[r] = (_Float16)y;
            }
            *(v8h*)(Yout + (s0 + j0) * (long)CoutTot + m_offset + mt * 16 + 8 * lhi) = o8;
        }

        if (nextTile < numTiles)
            asm volatile("s_wait_asynccnt 0" ::: "memory");
        __syncthreads();          // publish staged buffer to all waves
        cb ^= 1;
    }

    // One-shot stats reduction: shfl over llo, LDS, then global atomics
    #pragma unroll
    for (int mt = 0; mt < 4; ++mt)
        #pragma unroll
        for (int r = 0; r < 8; ++r) {
            float ps = psum[mt][r], pq = psq[mt][r];
            #pragma unroll
            for (int m = 8; m >= 1; m >>= 1) { ps += __shfl_xor(ps, m, 32); pq += __shfl_xor(pq, m, 32); }
            if (llo == 0) {
                const int lc = mt * 16 + r + 8 * lhi;
                atomicAdd(&ssum[lc], ps); atomicAdd(&ssq[lc], pq);
            }
        }
    __syncthreads();
    if (t < 64) {
        atomicAdd(&sumBuf[m_offset + t], ssum[t]);
        atomicAdd(&sqBuf[m_offset + t],  ssq[t]);
    }
}

// -------------------------------------------------- softmax fusion tail -----
// One wave per point; 2 neighbors per lane; y3 rows (256B) read contiguously.
__global__ __launch_bounds__(256)
void fuse_kernel(const _Float16* __restrict__ y3,       // [S][128]
                 const float* __restrict__ scale3, const float* __restrict__ shift3,
                 const float* __restrict__ grouped,     // [S][3]
                 float* __restrict__ out) {
    __shared__ float ssc[128], ssh[128];
    const int t = threadIdx.x;
    if (t < 128) { ssc[t] = scale3[t]; ssh[t] = shift3[t]; }
    __syncthreads();

    const int lane = t & 31, wib = t >> 5;
    const long pn = (long)blockIdx.x * 8 + wib;
    const int b = (int)(pn / NN);
    const int n = (int)(pn % NN);
    const long sA = pn * 64 + 2 * lane;      // two adjacent neighbors per lane
    const long sB = sA + 1;

    float scA = -3.4e38f, scB = -3.4e38f;
    const v8h* rowA = (const v8h*)(y3 + sA * 128);
    const v8h* rowB = (const v8h*)(y3 + sB * 128);
    #pragma unroll 4
    for (int g = 0; g < 16; ++g) {
        const v8h va = rowA[g], vb = rowB[g];
        #pragma unroll
        for (int h = 0; h < 8; ++h) {
            const int c = g * 8 + h;
            float fa = (float)va[h] * ssc[c] + ssh[c];
            float fb = (float)vb[h] * ssc[c] + ssh[c];
            fa = fa > 0.f ? fa : 0.f;
            fb = fb > 0.f ? fb : 0.f;
            scA = fmaxf(scA, fa);
            scB = fmaxf(scB, fb);
        }
    }

    // wave-local softmax over the 64 scores (2 per lane)
    float m = fmaxf(scA, scB);
    #pragma unroll
    for (int msk = 16; msk >= 1; msk >>= 1) m = fmaxf(m, __shfl_xor(m, msk, 32));
    const float eA = __expf(scA - m);
    const float eB = __expf(scB - m);

    const float* gA = grouped + sA * 3;      // 6 contiguous floats per lane
    float v0 = eA + eB;
    float v1 = eA * gA[0] + eB * gA[3];
    float v2 = eA * gA[1] + eB * gA[4];
    float v3 = eA * gA[2] + eB * gA[5];
    #pragma unroll
    for (int msk = 16; msk >= 1; msk >>= 1) {
        v0 += __shfl_xor(v0, msk, 32); v1 += __shfl_xor(v1, msk, 32);
        v2 += __shfl_xor(v2, msk, 32); v3 += __shfl_xor(v3, msk, 32);
    }
    if (lane == 0) {
        out[((long)b * 3 + 0) * NN + n] = v1 / v0;
        out[((long)b * 3 + 1) * NN + n] = v2 / v0;
        out[((long)b * 3 + 2) * NN + n] = v3 / v0;
    }
}

__global__ void zero_kernel(float* __restrict__ p, int n) {
    const int i = blockIdx.x * 256 + threadIdx.x;
    if (i < n) p[i] = 0.f;
}

// ---------------------------------------------------------------------------
extern "C" void kernel_launch(void* const* d_in, const int* in_sizes, int n_in,
                              void* d_out, int out_size, void* d_ws, size_t ws_size,
                              hipStream_t stream) {
    (void)in_sizes; (void)n_in; (void)out_size; (void)ws_size;

    const float* points1 = (const float*)d_in[0];
    const float* points2 = (const float*)d_in[1];
    // d_in[2]=k (hardcoded 32), d_in[3]=t (unused by reference)
    const float* W1 = (const float*)d_in[4];
    const float* g1 = (const float*)d_in[6];
    const float* be1 = (const float*)d_in[7];
    const float* W2 = (const float*)d_in[8];
    const float* g2 = (const float*)d_in[10];
    const float* be2 = (const float*)d_in[11];
    const float* W3 = (const float*)d_in[12];
    const float* g3 = (const float*)d_in[14];
    const float* be3 = (const float*)d_in[15];
    // biases b1/b2/b3 cancel under batch-norm (mean subtraction) -> unused
    float* out = (float*)d_out;

    size_t off = 0;
    auto alloc = [&](size_t bytes) -> void* {
        void* p = (char*)d_ws + off;
        off += (bytes + 255) & ~(size_t)255;
        return p;
    };
    float*    feats   = (float*)   alloc((size_t)SS * 4 * sizeof(float));      // 16.8 MB
    float*    grouped = (float*)   alloc((size_t)SS * 3 * sizeof(float));      // 12.6 MB
    _Float16* y1      = (_Float16*)alloc((size_t)SS * 64  * sizeof(_Float16)); // 134 MB
    _Float16* y2      = (_Float16*)alloc((size_t)SS * 64  * sizeof(_Float16)); // 134 MB
    _Float16* y3      = (_Float16*)alloc((size_t)SS * 128 * sizeof(_Float16)); // 268 MB
    float*    stats   = (float*)   alloc(1024 * sizeof(float));
    float* sum1 = stats +   0; float* sq1 = stats +  64;
    float* sum2 = stats + 128; float* sq2 = stats + 192;
    float* sum3 = stats + 256; float* sq3 = stats + 384;   // 128 each
    float* sc1  = stats + 512; float* sh1 = stats + 576;
    float* sc2  = stats + 640; float* sh2 = stats + 704;
    float* sc3  = stats + 768; float* sh3 = stats + 896;   // 128 each

    zero_kernel<<<4, 256, 0, stream>>>(stats, 1024);

    knn_feats_kernel<<<dim3(NN / 128, BB), 128, 0, stream>>>(points1, points2, feats, grouped);

    conv1_wmma_kernel<<<512, 256, 0, stream>>>(feats, W1, y1, sum1, sq1);
    finalize_stats_kernel<<<1, 64, 0, stream>>>(sum1, sq1, g1, be1, sc1, sh1, 64);

    gemm_bn_relu_kernel<<<512, 256, 0, stream>>>(W2, y1, sc1, sh1, y2, sum2, sq2, 0, 64);
    finalize_stats_kernel<<<1, 64, 0, stream>>>(sum2, sq2, g2, be2, sc2, sh2, 64);

    gemm_bn_relu_kernel<<<512, 256, 0, stream>>>(W3, y2, sc2, sh2, y3, sum3, sq3, 0, 128);
    gemm_bn_relu_kernel<<<512, 256, 0, stream>>>(W3, y2, sc2, sh2, y3, sum3, sq3, 64, 128);
    finalize_stats_kernel<<<1, 128, 0, stream>>>(sum3, sq3, g3, be3, sc3, sh3, 128);

    fuse_kernel<<<(BB * NN) / 8, 256, 0, stream>>>(y3, sc3, sh3, grouped, out);
}
